// KoLeoLoss_Triplet_9955734192549
// MI455X (gfx1250) — compile-verified, matrix operand order checked by reading, not run on hardware
//
#include <hip/hip_runtime.h>
#include <hip/hip_bf16.h>
#include <math.h>

typedef __attribute__((ext_vector_type(16))) __bf16   v16bf;
typedef __attribute__((ext_vector_type(8)))  float    v8f;
typedef __attribute__((ext_vector_type(4)))  unsigned v4u;

#define DDIM  512   // feature dim
#define BT    128   // block tile (rows == cols)
#define KC    32    // k-chunk per WMMA pass
#define NK    (DDIM / KC)   // 16 chunks
#define LDSTR 20    // dwords per LDS row: 16 k-pairs + pad (16B align, conflict-free banks)

// ---------- bf16 helpers (round-to-nearest-even split) ----------
__device__ __forceinline__ unsigned short f2bf(float f) {
    unsigned u = __float_as_uint(f);
    u += 0x7FFFu + ((u >> 16) & 1u);
    return (unsigned short)(u >> 16);
}
__device__ __forceinline__ float bf2f(unsigned short h) {
    return __uint_as_float(((unsigned)h) << 16);
}

union FragU { v4u q[2]; v16bf b; };

// Fragment gather matching the CDNA5 16-bit A/B 16x32 layout:
// lanes 0-15: row m, K-pairs {0..3, 8..11}; lanes 16-31: row m, K-pairs {4..7, 12..15}
__device__ __forceinline__ v16bf load_frag(const unsigned* base, int tileRow, int lane) {
    const int m    = lane & 15;
    const int half = lane >> 4;
    const unsigned* p = base + (tileRow + m) * LDSTR + half * 4;
    FragU f;
    f.q[0] = *(const v4u*)(p);       // K-pairs half*4 .. half*4+3
    f.q[1] = *(const v4u*)(p + 8);   // K-pairs half*4+8 .. half*4+11
    return f.b;
}

// ---------- kernel 1: init per-row min to +inf ----------
__global__ void koleo_init_min(int* __restrict__ rowMinBits, int n) {
    int i = blockIdx.x * 256 + threadIdx.x;
    if (i < n) rowMinBits[i] = 0x7F800000;  // +inf
}

// ---------- kernel 2: row norms (one wave per row) ----------
__global__ __launch_bounds__(256)
void koleo_row_norms(const float* __restrict__ a, const float* __restrict__ p,
                     const float* __restrict__ ng, float* __restrict__ norms, int n) {
    int gw   = (blockIdx.x * 256 + threadIdx.x) >> 5;   // row id in [0, 3n)
    int lane = threadIdx.x & 31;
    if (gw >= 3 * n) return;
    int mat = gw / n;
    int r   = gw - mat * n;
    const float* src = (mat == 0) ? a : (mat == 1) ? p : ng;
    const float* rp  = src + (size_t)r * DDIM;
    float s = 0.0f;
    #pragma unroll
    for (int i = 0; i < DDIM / 128; ++i) {
        float4 v = *(const float4*)(rp + lane * 4 + i * 128);
        s += v.x * v.x + v.y * v.y + v.z * v.z + v.w * v.w;
    }
    #pragma unroll
    for (int d = 16; d; d >>= 1) s += __shfl_xor(s, d, 32);
    if (lane == 0) norms[gw] = s;
}

// ---------- kernel 3: fused split-bf16 WMMA GEMM + distance + row-min ----------
__global__ __launch_bounds__(256)
void koleo_dist_min(const float* __restrict__ anchor,
                    const float* __restrict__ positive,
                    const float* __restrict__ negative,
                    const float* __restrict__ norms,   // [3n] anchor|positive|negative
                    int* __restrict__ rowMinBits,
                    int n) {
    // [buf][0]=Ahi [buf][1]=Alo [buf][2]=Bhi [buf][3]=Blo  (double-buffered ping-pong)
    __shared__ unsigned ldsBuf[2][4][BT * LDSTR];

    const int z = blockIdx.z;
    const float* Bmat = (z == 0) ? anchor : (z == 1) ? positive : negative;
    const int rowBase = blockIdx.y * BT;
    const int colBase = blockIdx.x * BT;
    const int tid  = threadIdx.x;
    const int lane = tid & 31;
    const int wave = tid >> 5;
    const int waveRow = wave >> 1;   // 0..3 -> 32-row strip
    const int waveCol = wave & 1;    // 0..1 -> 64-col strip

    v8f acc[2][4];
    #pragma unroll
    for (int rt = 0; rt < 2; ++rt)
        #pragma unroll
        for (int ct = 0; ct < 4; ++ct)
            acc[rt][ct] = (v8f)0.0f;

    const int ldR  = tid >> 3;   // 0..31 (row within pass)
    const int ldC4 = tid & 7;    // float4 index (2 k-pairs)

    float4 ar[4], br[4];  // in-flight global tile (A rows / B rows)

    // ---- helpers as lambdas (inlined) ----
    auto globLoad = [&](int kci) {
        const int kc = kci * KC;
        #pragma unroll
        for (int pass = 0; pass < 4; ++pass) {
            const int r = pass * 32 + ldR;
            ar[pass] = *(const float4*)(anchor + (size_t)(rowBase + r) * DDIM + kc + ldC4 * 4);
            br[pass] = *(const float4*)(Bmat   + (size_t)(colBase + r) * DDIM + kc + ldC4 * 4);
        }
    };
    auto stage = [&](int buf) {
        #pragma unroll
        for (int pass = 0; pass < 4; ++pass) {
            const int r = pass * 32 + ldR;
            {
                float4 v = ar[pass];
                unsigned short h0 = f2bf(v.x), h1 = f2bf(v.y), h2 = f2bf(v.z), h3 = f2bf(v.w);
                unsigned short l0 = f2bf(v.x - bf2f(h0)), l1 = f2bf(v.y - bf2f(h1));
                unsigned short l2 = f2bf(v.z - bf2f(h2)), l3 = f2bf(v.w - bf2f(h3));
                ldsBuf[buf][0][r * LDSTR + ldC4 * 2 + 0] = (unsigned)h0 | ((unsigned)h1 << 16);
                ldsBuf[buf][0][r * LDSTR + ldC4 * 2 + 1] = (unsigned)h2 | ((unsigned)h3 << 16);
                ldsBuf[buf][1][r * LDSTR + ldC4 * 2 + 0] = (unsigned)l0 | ((unsigned)l1 << 16);
                ldsBuf[buf][1][r * LDSTR + ldC4 * 2 + 1] = (unsigned)l2 | ((unsigned)l3 << 16);
            }
            {
                float4 v = br[pass];
                unsigned short h0 = f2bf(v.x), h1 = f2bf(v.y), h2 = f2bf(v.z), h3 = f2bf(v.w);
                unsigned short l0 = f2bf(v.x - bf2f(h0)), l1 = f2bf(v.y - bf2f(h1));
                unsigned short l2 = f2bf(v.z - bf2f(h2)), l3 = f2bf(v.w - bf2f(h3));
                ldsBuf[buf][2][r * LDSTR + ldC4 * 2 + 0] = (unsigned)h0 | ((unsigned)h1 << 16);
                ldsBuf[buf][2][r * LDSTR + ldC4 * 2 + 1] = (unsigned)h2 | ((unsigned)h3 << 16);
                ldsBuf[buf][3][r * LDSTR + ldC4 * 2 + 0] = (unsigned)l0 | ((unsigned)l1 << 16);
                ldsBuf[buf][3][r * LDSTR + ldC4 * 2 + 1] = (unsigned)l2 | ((unsigned)l3 << 16);
            }
        }
    };
    auto compute = [&](int buf) {
        v16bf aHi[2], aLo[2];
        #pragma unroll
        for (int rt = 0; rt < 2; ++rt) {
            aHi[rt] = load_frag(ldsBuf[buf][0], waveRow * 32 + rt * 16, lane);
            aLo[rt] = load_frag(ldsBuf[buf][1], waveRow * 32 + rt * 16, lane);
        }
        #pragma unroll
        for (int ct = 0; ct < 4; ++ct) {
            v16bf bHi = load_frag(ldsBuf[buf][2], waveCol * 64 + ct * 16, lane);
            v16bf bLo = load_frag(ldsBuf[buf][3], waveCol * 64 + ct * 16, lane);
            #pragma unroll
            for (int rt = 0; rt < 2; ++rt) {
                acc[rt][ct] = __builtin_amdgcn_wmma_f32_16x16x32_bf16(
                    false, aHi[rt], false, bHi, (short)0, acc[rt][ct], false, false);
                acc[rt][ct] = __builtin_amdgcn_wmma_f32_16x16x32_bf16(
                    false, aHi[rt], false, bLo, (short)0, acc[rt][ct], false, false);
                acc[rt][ct] = __builtin_amdgcn_wmma_f32_16x16x32_bf16(
                    false, aLo[rt], false, bHi, (short)0, acc[rt][ct], false, false);
            }
        }
    };

    // ---- software-pipelined ping-pong main loop ----
    globLoad(0);
    stage(0);
    __syncthreads();

    #pragma unroll 1
    for (int kci = 0; kci < NK; kci += 2) {
        // even chunk: compute buf0 while fetching/staging chunk kci+1 into buf1
        if (kci + 1 < NK) globLoad(kci + 1);
        compute(0);
        if (kci + 1 < NK) stage(1);
        __syncthreads();
        // odd chunk: compute buf1 while fetching/staging chunk kci+2 into buf0
        if (kci + 2 < NK) globLoad(kci + 2);
        if (kci + 1 < NK) compute(1);
        if (kci + 2 < NK) stage(0);
        __syncthreads();
    }

    // ---- epilogue: dist = sqrt(max(na+nb-2s,0)); mask self/zero; row-min ----
    const float* nb = norms + (size_t)z * n;
    const int mcol    = lane & 15;        // N within 16x16 tile
    const int mrowOff = (lane >> 4) * 8;  // lanes 16-31 hold M = v+8

    #pragma unroll
    for (int rt = 0; rt < 2; ++rt) {
        float rmin[8];
        #pragma unroll
        for (int v = 0; v < 8; ++v) rmin[v] = INFINITY;
        #pragma unroll
        for (int ct = 0; ct < 4; ++ct) {
            const int col  = colBase + waveCol * 64 + ct * 16 + mcol;
            const float nbv = nb[col];
            #pragma unroll
            for (int v = 0; v < 8; ++v) {
                const int row = rowBase + waveRow * 32 + rt * 16 + mrowOff + v;
                float sq = norms[row] + nbv - 2.0f * acc[rt][ct][v];
                sq = fmaxf(sq, 0.0f);
                const bool excl = (sq == 0.0f) || ((z == 0) && (row == col));
                const float dist = excl ? INFINITY : sqrtf(sq);
                rmin[v] = fminf(rmin[v], dist);
            }
        }
        #pragma unroll
        for (int v = 0; v < 8; ++v) {
            float m = rmin[v];
            m = fminf(m, __shfl_xor(m, 8, 16));
            m = fminf(m, __shfl_xor(m, 4, 16));
            m = fminf(m, __shfl_xor(m, 2, 16));
            m = fminf(m, __shfl_xor(m, 1, 16));
            if ((lane & 15) == 0) {
                const int row = rowBase + waveRow * 32 + rt * 16 + mrowOff + v;
                atomicMin(rowMinBits + row, __float_as_int(m));  // valid: all values >= 0
            }
        }
    }
}

// ---------- kernel 4: -mean(log(min_d)) ----------
__global__ __launch_bounds__(256)
void koleo_finalize(const int* __restrict__ rowMinBits, float* __restrict__ out, int n) {
    __shared__ float red[256];
    float s = 0.0f;
    for (int i = threadIdx.x; i < n; i += 256) {
        float m = __int_as_float(rowMinBits[i]);
        if (!(m > 0.0f) || m == INFINITY) m = 1.0f;  // degenerate-row guard
        s += logf(m);
    }
    red[threadIdx.x] = s;
    __syncthreads();
    for (int st = 128; st; st >>= 1) {
        if (threadIdx.x < st) red[threadIdx.x] += red[threadIdx.x + st];
        __syncthreads();
    }
    if (threadIdx.x == 0) out[0] = -red[0] / (float)n;
}

extern "C" void kernel_launch(void* const* d_in, const int* in_sizes, int n_in,
                              void* d_out, int out_size, void* d_ws, size_t ws_size,
                              hipStream_t stream) {
    const float* anchor   = (const float*)d_in[0];
    const float* positive = (const float*)d_in[1];
    const float* negative = (const float*)d_in[2];
    float* out = (float*)d_out;

    const int n = in_sizes[0] / DDIM;   // 8192

    int*   rowMinBits = (int*)d_ws;            // n ints
    float* norms      = (float*)d_ws + n;      // 3n floats

    koleo_init_min<<<(n + 255) / 256, 256, 0, stream>>>(rowMinBits, n);
    koleo_row_norms<<<(3 * n * 32 + 255) / 256, 256, 0, stream>>>(anchor, positive, negative, norms, n);

    dim3 grid(n / BT, n / BT, 3);
    koleo_dist_min<<<grid, 256, 0, stream>>>(anchor, positive, negative, norms, rowMinBits, n);

    koleo_finalize<<<1, 256, 0, stream>>>(rowMinBits, out, n);
}
